// MultiScaleRetention_39951785787714
// MI455X (gfx1250) — matrix operand order repaired and checked by reading, not compiled, not run on hardware
//
#include <hip/hip_runtime.h>
#include <hip/hip_bf16.h>

// ---------------- problem constants ----------------
#define BB 2
#define TT 2048
#define EE 2048
#define VV 4096
#define NHH 8
#define KDD 256   // EE/NHH
#define HDD 512   // VV/NHH
#define EPSF 1e-6f

typedef __attribute__((ext_vector_type(16))) __bf16 bf16x16;
typedef __attribute__((ext_vector_type(8)))  __bf16 bf16x8;
typedef __attribute__((ext_vector_type(8)))  float  f32x8;
typedef __attribute__((ext_vector_type(4)))  int    i32x4;

#define DEV static __device__ __forceinline__

// CDNA5 async global->LDS path (ASYNCcnt), guarded so the file compiles on
// toolchains where the builtins are absent (falls back to sync VGPR relay).
#if defined(__gfx1250__) && defined(__has_builtin)
# if __has_builtin(__builtin_amdgcn_global_load_async_to_lds_b128) && \
     __has_builtin(__builtin_amdgcn_s_wait_asynccnt)
#  define USE_ASYNC_LDS 1
# endif
#endif
#ifndef USE_ASYNC_LDS
# define USE_ASYNC_LDS 0
#endif

#if USE_ASYNC_LDS
typedef __attribute__((address_space(1))) i32x4 gas_i32x4;   // global int4
typedef __attribute__((address_space(3))) i32x4 las_i32x4;   // LDS int4
#endif

// copy 16 bytes global -> LDS (async when available)
DEV void cp16_g2l(const __bf16* g, __bf16* l) {
#if USE_ASYNC_LDS
  __builtin_amdgcn_global_load_async_to_lds_b128(
      (gas_i32x4*)(unsigned long long)g,
      (las_i32x4*)(unsigned int)(unsigned long long)l, 0, 0);
#else
  *(bf16x8*)l = *(const bf16x8*)g;
#endif
}

DEV void stage_wait() {
#if USE_ASYNC_LDS
  __builtin_amdgcn_s_wait_asynccnt(0);
#endif
}

DEV f32x8 fzero8() {
  f32x8 z = {0.f, 0.f, 0.f, 0.f, 0.f, 0.f, 0.f, 0.f};
  return z;
}

// float -> bf16, round-to-nearest-even
DEV __bf16 f2bf(float f) {
  union { float f; unsigned u; } v; v.f = f;
  unsigned u = v.u;
  unsigned r = u + 0x7FFFu + ((u >> 16) & 1u);
  unsigned short h = (unsigned short)(r >> 16);
  __bf16 b;
  __builtin_memcpy(&b, &h, 2);
  return b;
}

// A fragment: 16x32 bf16 (MxK), src row-major, leading dim ld (elements).
// ISA 7.12.2: lanes 0-15 row M=lane, K 0..7 -> elems 0..7, K 16..23 -> 8..15;
// lanes 16-31 same rows, K 8..15 / 24..31.
DEV bf16x16 load_a16x32(const __bf16* base, int ld) {
  int lane = threadIdx.x & 31;
  int r = lane & 15;
  int g = (lane >> 4) << 3;           // 0 or 8
  const __bf16* p = base + (size_t)r * ld + g;
  bf16x8 lo = *(const bf16x8*)(p);
  bf16x8 hi = *(const bf16x8*)(p + 16);
  bf16x16 out;
#pragma unroll
  for (int i = 0; i < 8; ++i) { out[i] = lo[i]; out[i + 8] = hi[i]; }
  return out;
}

// B fragment: 32x16 bf16 (KxN) with B[k][n] = W[n][k], W row-major NxK.
// lane n = lane&15; lanes 0-15 hold K 0..7 / 16..23, lanes 16-31 hold K 8..15 / 24..31.
DEV bf16x16 load_bT32x16(const __bf16* base, int ld) {
  int lane = threadIdx.x & 31;
  int n = lane & 15;
  int g = (lane >> 4) << 3;           // 0 or 8
  const __bf16* p = base + (size_t)n * ld + g;
  bf16x8 lo = *(const bf16x8*)(p);
  bf16x8 hi = *(const bf16x8*)(p + 16);
  bf16x16 out;
#pragma unroll
  for (int i = 0; i < 8; ++i) { out[i] = lo[i]; out[i + 8] = hi[i]; }
  return out;
}

DEV f32x8 wmma_bf16(bf16x16 a, bf16x16 b, f32x8 c) {
  return __builtin_amdgcn_wmma_f32_16x16x32_bf16(
      false, a, false, b, (short)0, c, false, false);
}

// ---------------- elementwise kernels ----------------
__global__ void k_cvt_bf16(const float* __restrict__ src,
                           __bf16* __restrict__ dst, int n) {
  for (int i = blockIdx.x * blockDim.x + threadIdx.x; i < n;
       i += gridDim.x * blockDim.x)
    dst[i] = f2bf(src[i]);
}

// rotary (theta-shift) on (B*T, E) f32 -> bf16, scale folded in
__global__ void k_rotary(const float* __restrict__ src,
                         __bf16* __restrict__ dst,
                         const float* __restrict__ sinp,
                         const float* __restrict__ cosp,
                         float scale) {
  const int total = BB * TT * (EE / 2);
  for (int i = blockIdx.x * blockDim.x + threadIdx.x; i < total;
       i += gridDim.x * blockDim.x) {
    int row = i >> 10;                 // E/2 = 1024 pairs per row
    int pc  = i & 1023;
    int c0  = pc << 1;
    int kd0 = c0 & (KDD - 1);
    int t   = row & (TT - 1);
    size_t base = (size_t)row * EE + c0;
    float x0 = src[base]     * scale;
    float x1 = src[base + 1] * scale;
    float s0 = sinp[(size_t)t * KDD + kd0];
    float s1 = sinp[(size_t)t * KDD + kd0 + 1];
    float c0v = cosp[(size_t)t * KDD + kd0];
    float c1v = cosp[(size_t)t * KDD + kd0 + 1];
    dst[base]     = f2bf(x0 * c0v - x1 * s0);
    dst[base + 1] = f2bf(x1 * c1v + x0 * s1);
  }
}

// v (B*T, V) f32 -> vT (B, NH, HD, T) bf16
__global__ void k_transpose_v(const float* __restrict__ v,
                              __bf16* __restrict__ vT) {
  const int total = BB * TT * VV;
  for (int i = blockIdx.x * blockDim.x + threadIdx.x; i < total;
       i += gridDim.x * blockDim.x) {
    int t = i & (TT - 1);
    int rest = i >> 11;                // /T
    int d = rest & (HDD - 1);
    rest >>= 9;                        // /HD
    int h = rest & (NHH - 1);
    int b = rest >> 3;                 // /NH
    vT[i] = f2bf(v[((size_t)(b * TT + t)) * VV + h * HDD + d]);
  }
}

// ------- C(M,N) = A(M,K) @ W(N,K)^T, bf16 in, f32 out -----------------------
// Block 256 threads (8 waves, 4x2). Tile 64x128, K-step 32, double-buffered
// LDS staging (A 4KB + B 8KB per buffer) fed by async global->LDS copies.
__global__ __launch_bounds__(256)
void k_gemm_xwT(const __bf16* __restrict__ A, const __bf16* __restrict__ W,
                float* __restrict__ C, int M, int N, int K) {
  __shared__ __align__(16) __bf16 LdsA[2][64 * 32];
  __shared__ __align__(16) __bf16 LdsB[2][128 * 32];
  (void)M;
  const int t  = threadIdx.x;
  const int w  = t >> 5;
  const int wm = w >> 1, wn = w & 1;           // 4x2 wave grid
  const int i0b = blockIdx.y * 64;
  const int n0b = blockIdx.x * 128;

  // staging map: A = 256 16B-chunks (64x32), B = 512 chunks (128x32);
  // every thread moves 1 A-chunk + 2 B-chunks per K-step (no divergence).
  const int ar  = t >> 2, ac = (t & 3) << 3;   // A row, col(elems)
  const __bf16* gA = A + (size_t)(i0b + ar) * K + ac;
  __bf16* lA0 = &LdsA[0][t << 3];
  const int br0 = t >> 2,         bc0 = (t & 3) << 3;
  const int br1 = (t + 256) >> 2, bc1 = (t & 3) << 3;
  const __bf16* gB0 = W + (size_t)(n0b + br0) * K + bc0;
  const __bf16* gB1 = W + (size_t)(n0b + br1) * K + bc1;
  __bf16* lB0 = &LdsB[0][t << 3];
  __bf16* lB1 = &LdsB[0][(t + 256) << 3];
  const size_t ldsStrideA = 64 * 32;           // elems between buffers
  const size_t ldsStrideB = 128 * 32;

  f32x8 acc[4];
#pragma unroll
  for (int nt = 0; nt < 4; ++nt) acc[nt] = fzero8();

  const int nk = K >> 5;
  // prologue: stage K-tile 0 into buffer 0
  cp16_g2l(gA, lA0);
  cp16_g2l(gB0, lB0);
  cp16_g2l(gB1, lB1);
  stage_wait();
  __syncthreads();

  for (int s = 0; s < nk; ++s) {
    const int cur = s & 1;
    if (s + 1 < nk) {                          // stream next tile into buf^1
      const int k1 = (s + 1) << 5;
      const int nxt = cur ^ 1;
      __builtin_prefetch(gA + k1 + 64, 0, 3);  // k+2 tile -> global_prefetch_b8
      cp16_g2l(gA + k1,  lA0 + nxt * ldsStrideA);
      cp16_g2l(gB0 + k1, lB0 + nxt * ldsStrideB);
      cp16_g2l(gB1 + k1, lB1 + nxt * ldsStrideB);
    }
    // compute from current buffer (ds_load_b128 fragments)
    bf16x16 a = load_a16x32(&LdsA[cur][wm * 16 * 32], 32);
#pragma unroll
    for (int nt = 0; nt < 4; ++nt) {
      bf16x16 b = load_bT32x16(&LdsB[cur][(wn * 64 + nt * 16) * 32], 32);
      acc[nt] = wmma_bf16(a, b, acc[nt]);
    }
    stage_wait();                              // my async fills landed
    __syncthreads();                           // everyone done reading `cur`
  }

  const int lane = t & 31;
  const int n = lane & 15;
  const int mb = (lane >> 4) << 3;
#pragma unroll
  for (int nt = 0; nt < 4; ++nt)
#pragma unroll
    for (int e = 0; e < 8; ++e)
      C[(size_t)(i0b + wm * 16 + mb + e) * N + n0b + wn * 64 + nt * 16 + n] =
          acc[nt][e];
}

// ---------------- fused retention + denom + rmsnorm + swish-gate ------------
// grid: (T/16, NH, B), block: 256 threads = 8 waves.
// Block owns one (b,h) and 16 query rows. Wave w builds S sub-tile at
// j-offset 16w, then consumes the shared 16x128 S tile against its 64-wide
// HD slice of vT. Epilogue fuses 1/denom, per-head RMS norm, swish(g) gate.
__global__ __launch_bounds__(256)
void k_retention(const __bf16* __restrict__ qr, const __bf16* __restrict__ kr,
                 const __bf16* __restrict__ vT, const float* __restrict__ gf,
                 const float* __restrict__ decay, __bf16* __restrict__ gated) {
  __shared__ __align__(16) __bf16 Sld[16 * 128];
  __shared__ float red[32];            // [0:16) abs-sum, [16:32) sum-of-squares

  const int i0 = blockIdx.x * 16;
  const int h  = blockIdx.y;
  const int b  = blockIdx.z;
  const int w    = threadIdx.x >> 5;
  const int lane = threadIdx.x & 31;
  const int n    = lane & 15;
  const int mb   = (lane >> 4) << 3;   // 0 or 8 (row base this lane covers)

  if (threadIdx.x < 32) red[threadIdx.x] = 0.f;

  // qr A-fragments held resident: 16 rows x KD=256 (8 k-steps)
  const __bf16* qbase = qr + (size_t)(b * TT + i0) * EE + h * KDD;
  bf16x16 qa[8];
#pragma unroll
  for (int kk = 0; kk < 8; ++kk) qa[kk] = load_a16x32(qbase + kk * 32, EE);

  const int d0 = w * 64;               // this wave's HD slice
  const __bf16* vTb = vT + ((size_t)((b * NHH + h) * HDD + d0)) * TT;
  const float* dmb = decay + ((size_t)(h * TT + i0 + mb)) * TT;

  f32x8 acc[4];
#pragma unroll
  for (int nt = 0; nt < 4; ++nt) acc[nt] = fzero8();
  float absp[8];
#pragma unroll
  for (int e = 0; e < 8; ++e) absp[e] = 0.f;

  for (int j0 = 0; j0 <= i0; j0 += 128) {
    const int jw = j0 + w * 16;
    // ---- S phase: S(16x16) = qr_i @ kr_jw^T over KD=256 ----
    f32x8 s = fzero8();
    const __bf16* kbase = kr + (size_t)(b * TT + jw) * EE + h * KDD;
#pragma unroll
    for (int kk = 0; kk < 8; ++kk) {
      bf16x16 kb = load_bT32x16(kbase + kk * 32, EE);
      s = wmma_bf16(qa[kk], kb, s);
    }
    __syncthreads();                   // prior iteration's S reads complete
#pragma unroll
    for (int e = 0; e < 8; ++e) {
      float dv = dmb[(size_t)e * TT + jw + n];   // decay (0 above diagonal)
      float sv = s[e] * dv;
      absp[e] += fabsf(sv);
      Sld[(mb + e) * 128 + (w << 4) + n] = f2bf(sv);
    }
    __syncthreads();                   // S tile visible to all waves
    // ---- out phase: acc += S(16x128) @ v(128 x 64-slice) ----
#pragma unroll
    for (int kk = 0; kk < 4; ++kk) {
      bf16x16 sa = load_a16x32(&Sld[kk * 32], 128);
#pragma unroll
      for (int nt = 0; nt < 4; ++nt) {
        bf16x16 vb = load_bT32x16(vTb + (size_t)(nt * 16) * TT + j0 + kk * 32, TT);
        acc[nt] = wmma_bf16(sa, vb, acc[nt]);
      }
    }
  }

  // ---- denom = clip(sum_j |S|, 1, 50000): lanes (shfl) then waves (LDS) ----
#pragma unroll
  for (int e = 0; e < 8; ++e)
#pragma unroll
    for (int m = 1; m < 16; m <<= 1)
      absp[e] += __shfl_xor(absp[e], m, 16);
  if (n == 0) {                        // lanes 0 and 16 of each wave
#pragma unroll
    for (int e = 0; e < 8; ++e) atomicAdd(&red[mb + e], absp[e]);
  }
  __syncthreads();

  float sqp[8];
#pragma unroll
  for (int e = 0; e < 8; ++e) {
    float dsum = fminf(fmaxf(red[mb + e], 1.f), 50000.f);
    float inv = 1.f / dsum;
    float sq = 0.f;
#pragma unroll
    for (int nt = 0; nt < 4; ++nt) {
      float o = acc[nt][e] * inv;
      acc[nt][e] = o;
      sq += o * o;
    }
    sqp[e] = sq;
  }
  // ---- RMS norm over HD=512 (this block covers the full head) ----
#pragma unroll
  for (int e = 0; e < 8; ++e)
#pragma unroll
    for (int m = 1; m < 16; m <<= 1)
      sqp[e] += __shfl_xor(sqp[e], m, 16);
  if (n == 0) {
#pragma unroll
    for (int e = 0; e < 8; ++e) atomicAdd(&red[16 + mb + e], sqp[e]);
  }
  __syncthreads();

#pragma unroll
  for (int e = 0; e < 8; ++e) {
    float rms = rsqrtf(red[16 + mb + e] * (1.f / (float)HDD) + EPSF);
    int row = i0 + mb + e;
    size_t rbase = (size_t)(b * TT + row) * VV + h * HDD + d0 + n;
#pragma unroll
    for (int nt = 0; nt < 4; ++nt) {
      float gv = gf[rbase + nt * 16];
      float sw = gv / (1.f + __expf(-gv));       // swish
      gated[rbase + nt * 16] = f2bf(acc[nt][e] * rms * sw);
    }
  }
}

// ---------------- host launch ----------------
extern "C" void kernel_launch(void* const* d_in, const int* in_sizes, int n_in,
                              void* d_out, int out_size, void* d_ws, size_t ws_size,
                              hipStream_t stream) {
  (void)in_sizes; (void)n_in; (void)out_size; (void)ws_size;
  const float* x    = (const float*)d_in[0];
  const float* sinp = (const float*)d_in[1];
  const float* cosp = (const float*)d_in[2];
  const float* dm   = (const float*)d_in[3];
  const float* Wq   = (const float*)d_in[4];
  const float* Wk   = (const float*)d_in[5];
  const float* Wv   = (const float*)d_in[6];
  const float* Wg   = (const float*)d_in[7];
  const float* Wo   = (const float*)d_in[8];
  float* out = (float*)d_out;

  char* ws = (char*)d_ws;
  size_t off = 0;
  auto carve = [&](size_t bytes) -> char* {
    char* p = ws + off;
    off += (bytes + 255) & ~(size_t)255;
    return p;
  };
  const size_t MT = (size_t)BB * TT;                 // 4096 rows
  __bf16* xb    = (__bf16*)carve(MT * EE * 2);
  __bf16* wqb   = (__bf16*)carve((size_t)EE * EE * 2);
  __bf16* wkb   = (__bf16*)carve((size_t)EE * EE * 2);
  __bf16* wvb   = (__bf16*)carve((size_t)VV * EE * 2);
  __bf16* wgb   = (__bf16*)carve((size_t)VV * EE * 2);
  __bf16* wob   = (__bf16*)carve((size_t)EE * VV * 2);
  float*  qf    = (float*)carve(MT * EE * 4);
  float*  kf    = (float*)carve(MT * EE * 4);
  __bf16* qrb   = (__bf16*)carve(MT * EE * 2);
  __bf16* krb   = (__bf16*)carve(MT * EE * 2);
  float*  vf    = (float*)carve(MT * VV * 4);
  float*  gf    = (float*)carve(MT * VV * 4);
  __bf16* vTb   = (__bf16*)carve(MT * VV * 2);
  __bf16* gated = (__bf16*)carve(MT * VV * 2);

  // 1) f32 -> bf16 conversions (activations + weights)
  k_cvt_bf16<<<4096, 256, 0, stream>>>(x,  xb,  (int)(MT * EE));
  k_cvt_bf16<<<4096, 256, 0, stream>>>(Wq, wqb, EE * EE);
  k_cvt_bf16<<<4096, 256, 0, stream>>>(Wk, wkb, EE * EE);
  k_cvt_bf16<<<4096, 256, 0, stream>>>(Wv, wvb, VV * EE);
  k_cvt_bf16<<<4096, 256, 0, stream>>>(Wg, wgb, VV * EE);
  k_cvt_bf16<<<4096, 256, 0, stream>>>(Wo, wob, EE * VV);

  // 2) projections: C = X @ W^T  (WMMA bf16 -> f32, async-LDS pipelined)
  dim3 blk(256);
  k_gemm_xwT<<<dim3(EE / 128, (int)(MT / 64)), blk, 0, stream>>>(xb, wqb, qf, (int)MT, EE, EE);
  k_gemm_xwT<<<dim3(EE / 128, (int)(MT / 64)), blk, 0, stream>>>(xb, wkb, kf, (int)MT, EE, EE);
  k_gemm_xwT<<<dim3(VV / 128, (int)(MT / 64)), blk, 0, stream>>>(xb, wvb, vf, (int)MT, VV, EE);
  k_gemm_xwT<<<dim3(VV / 128, (int)(MT / 64)), blk, 0, stream>>>(xb, wgb, gf, (int)MT, VV, EE);

  // 3) rotary on q and k (k pre-scaled by KD^-0.5), f32 -> bf16
  k_rotary<<<4096, 256, 0, stream>>>(qf, qrb, sinp, cosp, 1.0f);
  k_rotary<<<4096, 256, 0, stream>>>(kf, krb, sinp, cosp, 0.0625f); // 256^-0.5

  // 4) v -> vT (B,NH,HD,T) bf16 so retention's 2nd GEMM has contiguous B-frags
  k_transpose_v<<<8192, 256, 0, stream>>>(vf, vTb);

  // 5) fused flash-style retention + denom + rmsnorm + swish gate
  k_retention<<<dim3(TT / 16, NHH, BB), blk, 0, stream>>>(qrb, krb, vTb, gf, dm, gated);

  // 6) final projection: out = gated @ Wo^T  (f32 output)
  k_gemm_xwT<<<dim3(EE / 128, (int)(MT / 64)), blk, 0, stream>>>(gated, wob, out, (int)MT, EE, VV);
}
